// MultiHeadAttention_5952824672524
// MI455X (gfx1250) — matrix operand order repaired and checked by reading, not compiled, not run on hardware
//
#include <hip/hip_runtime.h>

#define DEV __device__ __forceinline__

typedef __attribute__((ext_vector_type(16))) __bf16 v16bf;
typedef __attribute__((ext_vector_type(8)))  float  v8f;
typedef __attribute__((ext_vector_type(4)))  unsigned int u32x4;

union FragU { v16bf v; u32x4 q[2]; };

// ---- problem constants -----------------------------------------------------
constexpr int Bb    = 2;
constexpr int T     = 2048;
constexpr int D     = 1024;
constexpr int H     = 16;
constexpr int DH    = 64;
constexpr int Mrows = Bb * T;   // 4096
constexpr int WAVES = 8;        // waves per 256-thread block (wave32)

// ---- helpers ---------------------------------------------------------------
DEV unsigned short f2bf(float f) {              // f32 -> bf16 (RNE)
  unsigned int u = __float_as_uint(f);
  u += 0x7FFFu + ((u >> 16) & 1u);
  return (unsigned short)(u >> 16);
}

DEV v8f wmma_bf16(v16bf a, v16bf b, v8f c) {
  return __builtin_amdgcn_wmma_f32_16x16x32_bf16(false, a, false, b,
                                                 (short)0, c, false, false);
}

// A fragment 16x32 from row-major bf16: lane m=lane&15 (row), hh=lane>>4.
// per ISA: element e -> K = e + 8*(e>=8) + 8*hh  => chunks at kk+8hh, kk+16+8hh
DEV v16bf load_a(const unsigned short* base, long stride, int row_base, int kk, int lane) {
  const int m = lane & 15, hh = lane >> 4;
  const unsigned short* p = base + (long)(row_base + m) * stride + kk + 8 * hh;
  FragU f;
  f.q[0] = *(const u32x4*)(p);
  f.q[1] = *(const u32x4*)(p + 16);
  return f.v;
}

// B fragment 32x16 from column-major B (== row-major B^T): lane n=lane&15 (col).
// per ISA: element e -> K = e + 8*(e>=8) + 8*hh  => chunks at kk+8hh, kk+16+8hh
DEV v16bf load_b(const unsigned short* base, long stride, int col_base, int kk, int lane) {
  const int n = lane & 15, hh = lane >> 4;
  const unsigned short* p = base + (long)(col_base + n) * stride + kk + 8 * hh;
  FragU f;
  f.q[0] = *(const u32x4*)(p);
  f.q[1] = *(const u32x4*)(p + 16);
  return f.v;
}

// ---- conversion kernels ----------------------------------------------------
__global__ void cvt_f32_bf16(const float* __restrict__ in,
                             unsigned short* __restrict__ out, int n) {
  int i = blockIdx.x * blockDim.x + threadIdx.x;
  if (i < n) out[i] = f2bf(in[i]);
}

// Wt[n*D + k] = W[k*D + n]   (bf16)
__global__ void transpose_f32_bf16(const float* __restrict__ W,
                                   unsigned short* __restrict__ Wt) {
  int i = blockIdx.x * blockDim.x + threadIdx.x;   // over D*D
  int n = i >> 10, k = i & (D - 1);
  Wt[i] = f2bf(W[(long)k * D + n]);
}

// ---- WMMA GEMM: C[M,D] = A[M,D] (bf16 rm) x Bt[D,D] (bf16 = W^T rm) --------
// Block = 8 waves sharing one 16-row A panel (staged to LDS via async copy);
// wave computes a 16x64 tile.  MODE 0: bf16 rm  MODE 1: bf16 V^T [b,h,dh,t]
// MODE 2: f32 + bias (final projection).
template <int MODE>
__global__ __launch_bounds__(256) void gemm_wmma(const unsigned short* __restrict__ A,
                                                 const unsigned short* __restrict__ Bt,
                                                 unsigned short* __restrict__ Ob,
                                                 float* __restrict__ Of,
                                                 const float* __restrict__ bias) {
  __shared__ unsigned short ldsA[16 * D];          // 32 KB shared A panel

  const int lane = threadIdx.x & 31;
  const int wave = threadIdx.x >> 5;
  const int mbase = (blockIdx.x >> 1) * 16;                  // shared by block
  const int nbase = (((blockIdx.x & 1) << 3) + wave) * 64;   // per-wave 64 cols

  // ---- stage A panel (16 rows x 1024 K, contiguous 32 KB) into LDS --------
  {
    const char* gsrc = (const char*)(A + (long)mbase * D);
    const unsigned lbase = (unsigned)(unsigned long long)(const void*)&ldsA[0];
#pragma unroll
    for (int t = 0; t < 8; ++t) {
      const int cb = ((int)threadIdx.x + t * 256) * 16;      // chunk byte offset
      const unsigned lo = lbase + (unsigned)cb;
      const unsigned long long ga = (unsigned long long)(gsrc + cb);
      asm volatile("global_load_async_to_lds_b128 %0, %1, off"
                   :: "v"(lo), "v"(ga) : "memory");
    }
    asm volatile("s_wait_asynccnt 0x0" ::: "memory");
    __syncthreads();
  }

  v8f acc[4] = {};
  for (int kk = 0; kk < D; kk += 32) {
    v16bf a = load_a(ldsA, D, 0, kk, lane);                  // ds_load_b128 x2
#pragma unroll
    for (int j = 0; j < 4; ++j) {
      v16bf b = load_b(Bt, D, nbase + j * 16, kk, lane);     // global_load_b128
      acc[j] = wmma_bf16(a, b, acc[j]);
    }
  }

  const int hh = lane >> 4, n0 = lane & 15;
#pragma unroll
  for (int j = 0; j < 4; ++j) {
#pragma unroll
    for (int e = 0; e < 8; ++e) {
      const int row = mbase + e + 8 * hh;      // C layout: M = vgpr + 8*half
      const int col = nbase + j * 16 + n0;     // N = lane&15
      const float v = acc[j][e];
      if (MODE == 0) {
        Ob[(long)row * D + col] = f2bf(v);
      } else if (MODE == 1) {                  // V^T: [b, h, dh, t]
        const int h = col >> 6, dh = col & 63;
        const int b_ = row >> 11, t = row & (T - 1);
        Ob[((long)(b_ * H + h) * DH + dh) * T + t] = f2bf(v);
      } else {
        Of[(long)row * D + col] = v + bias[col];
      }
    }
  }
}

// ---- flash attention: per-wave 16 query rows, 32-key tiles -----------------
__global__ __launch_bounds__(256) void attn_wmma(const unsigned short* __restrict__ Qb,
                                                 const unsigned short* __restrict__ Kb,
                                                 const unsigned short* __restrict__ Vt,
                                                 unsigned short* __restrict__ Ctx) {
  __shared__ unsigned short ldsP[WAVES][16 * 32];   // per-wave P tile (8 KB)

  const int lane = threadIdx.x & 31;
  const int wave = threadIdx.x >> 5;
  const int qblk = blockIdx.x & 15;                 // T/128 = 16
  const int bh   = blockIdx.x >> 4;
  const int b    = bh >> 4;                         // H = 16
  const int h    = bh & 15;
  const int qbase = qblk * 128 + wave * 16;
  const int hh = lane >> 4, ln = lane & 15;

  const float scale = 0.125f;                       // 1/sqrt(DH)
  const float slope = exp2f(-0.5f * (float)(h + 1));// ALiBi m_h = 2^(-8(h+1)/H)

  const unsigned short* Qh = Qb + (long)b * T * D + h * DH;
  const unsigned short* Kh = Kb + (long)b * T * D + h * DH;
  const unsigned short* Vh = Vt + (long)(b * H + h) * DH * T;

  const v16bf qa0 = load_a(Qh, D, qbase, 0,  lane); // dh 0..31
  const v16bf qa1 = load_a(Qh, D, qbase, 32, lane); // dh 32..63

  v8f ctx[4] = {};
  float mrun[8], lrun[8];
#pragma unroll
  for (int e = 0; e < 8; ++e) { mrun[e] = -1e30f; lrun[e] = 0.f; }

  unsigned short* myP = &ldsP[wave][0];
  const int nkt = (qbase + 47) >> 5;                // causal: keys <= qbase+15

  for (int kt = 0; kt < nkt; ++kt) {
    const int k0 = kt * 32;

    // S = Q K^T  (two 16-col subtiles, K-dim = DH = 64 chained)
    v8f s0 = {}, s1 = {};
    {
      v16bf kb;
      kb = load_b(Kh, D, k0,      0,  lane);  s0 = wmma_bf16(qa0, kb, s0);
      kb = load_b(Kh, D, k0,      32, lane);  s0 = wmma_bf16(qa1, kb, s0);
      kb = load_b(Kh, D, k0 + 16, 0,  lane);  s1 = wmma_bf16(qa0, kb, s1);
      kb = load_b(Kh, D, k0 + 16, 32, lane);  s1 = wmma_bf16(qa1, kb, s1);
    }

    // mask -> +alibi -> *1/sqrt(DH), online softmax, P -> LDS (bf16, 16x32 rm)
    const int j0 = k0 + ln, j1 = k0 + 16 + ln;
#pragma unroll
    for (int e = 0; e < 8; ++e) {
      const int i = qbase + e + 8 * hh;
      const bool ok0 = (j0 <= i), ok1 = (j1 <= i);
      const float v0 = ok0 ? (s0[e] + slope * (float)(j0 - i)) * scale : -1e30f;
      const float v1 = ok1 ? (s1[e] + slope * (float)(j1 - i)) * scale : -1e30f;

      float rmax = fmaxf(v0, v1);
#pragma unroll
      for (int d = 1; d < 16; d <<= 1) rmax = fmaxf(rmax, __shfl_xor(rmax, d, 32));
      const float mnew = fmaxf(mrun[e], rmax);
      const float corr = __expf(mrun[e] - mnew);
      mrun[e] = mnew;

      const float p0 = ok0 ? __expf(v0 - mnew) : 0.f;
      const float p1 = ok1 ? __expf(v1 - mnew) : 0.f;
      float rsum = p0 + p1;
#pragma unroll
      for (int d = 1; d < 16; d <<= 1) rsum += __shfl_xor(rsum, d, 32);
      lrun[e] = lrun[e] * corr + rsum;

#pragma unroll
      for (int j = 0; j < 4; ++j) ctx[j][e] = ctx[j][e] * corr;

      myP[(e + 8 * hh) * 32 + ln]      = f2bf(p0);
      myP[(e + 8 * hh) * 32 + 16 + ln] = f2bf(p1);
    }

    // wave-private LDS round trip: fence stores before A-layout reload
    asm volatile("s_wait_dscnt 0" ::: "memory");
    const v16bf pa = load_a(myP, 32, 0, 0, lane);

    // ctx += P V   (V^T rows are dh, contiguous over t)
#pragma unroll
    for (int j = 0; j < 4; ++j) {
      v16bf vb = load_b(Vh, T, j * 16, k0, lane);
      ctx[j] = wmma_bf16(pa, vb, ctx[j]);
    }
  }

  // normalize and store context (bf16 row-major [b*T, D], col = h*64+dh)
#pragma unroll
  for (int j = 0; j < 4; ++j) {
#pragma unroll
    for (int e = 0; e < 8; ++e) {
      const int row = qbase + e + 8 * hh;
      const float o = ctx[j][e] / lrun[e];
      Ctx[(long)(b * T + row) * D + h * DH + j * 16 + ln] = f2bf(o);
    }
  }
}

// ---- host launcher ---------------------------------------------------------
extern "C" void kernel_launch(void* const* d_in, const int* in_sizes, int n_in,
                              void* d_out, int out_size, void* d_ws, size_t ws_size,
                              hipStream_t stream) {
  const float* x  = (const float*)d_in[0];
  const float* Wq = (const float*)d_in[1];
  const float* Wk = (const float*)d_in[2];
  const float* Wv = (const float*)d_in[3];
  const float* Wp = (const float*)d_in[4];
  const float* bp = (const float*)d_in[5];
  float* out = (float*)d_out;

  char* ws = (char*)d_ws;
  const size_t MB = 1024u * 1024u;
  unsigned short* xb   = (unsigned short*)(ws + 0);        // 8 MB, reused as ctx
  unsigned short* Qb   = (unsigned short*)(ws + 8 * MB);   // 8 MB
  unsigned short* Kb   = (unsigned short*)(ws + 16 * MB);  // 8 MB
  unsigned short* Vt   = (unsigned short*)(ws + 24 * MB);  // 8 MB
  unsigned short* WqT  = (unsigned short*)(ws + 32 * MB);  // 2 MB
  unsigned short* WkT  = (unsigned short*)(ws + 34 * MB);  // 2 MB
  unsigned short* WvT  = (unsigned short*)(ws + 36 * MB);  // 2 MB
  unsigned short* WpT  = (unsigned short*)(ws + 38 * MB);  // 2 MB
  unsigned short* Ctxb = xb;                               // x dead after QKV

  const int nX = Mrows * D;          // 4,194,304
  const int nW = D * D;              // 1,048,576
  cvt_f32_bf16<<<(nX + 255) / 256, 256, 0, stream>>>(x, xb, nX);
  transpose_f32_bf16<<<(nW + 255) / 256, 256, 0, stream>>>(Wq, WqT);
  transpose_f32_bf16<<<(nW + 255) / 256, 256, 0, stream>>>(Wk, WkT);
  transpose_f32_bf16<<<(nW + 255) / 256, 256, 0, stream>>>(Wv, WvT);
  transpose_f32_bf16<<<(nW + 255) / 256, 256, 0, stream>>>(Wp, WpT);

  const int gblk = (Mrows / 16) * (D / 64) / WAVES;        // 512 blocks
  gemm_wmma<0><<<gblk, 256, 0, stream>>>(xb, WqT, Qb, nullptr, nullptr);
  gemm_wmma<0><<<gblk, 256, 0, stream>>>(xb, WkT, Kb, nullptr, nullptr);
  gemm_wmma<1><<<gblk, 256, 0, stream>>>(xb, WvT, Vt, nullptr, nullptr);

  attn_wmma<<<Bb * H * (T / 128), 256, 0, stream>>>(Qb, Kb, Vt, Ctxb);

  gemm_wmma<2><<<gblk, 256, 0, stream>>>(Ctxb, WpT, nullptr, out, bp);
}
